// Encoder_816043786373
// MI455X (gfx1250) — matrix operand order repaired and checked by reading, not compiled, not run on hardware
//
#include <hip/hip_runtime.h>

// ---------------------------------------------------------------------------
// FlowNetC-style encoder for MI455X (gfx1250, wave32, WMMA + TDM).
// Convs = implicit GEMM on v_wmma_f32_16x16x32_bf16 (fp32 accum):
//   - A (weight) tile staged to LDS by the Tensor Data Mover (tensor_load_to_lds,
//     D# pad feature reproduces the LDS bank-conflict padding, TENSORcnt sync)
//   - B (im2col) tile gathered/converted by the waves
//   - double-buffered LDS, one barrier per K-chunk, batch on blockIdx.z
// ---------------------------------------------------------------------------

typedef __attribute__((ext_vector_type(16))) __bf16 v16bf;
typedef __attribute__((ext_vector_type(8)))  float  v8f;
typedef unsigned int u32x4 __attribute__((ext_vector_type(4)));
typedef int          i32x4 __attribute__((ext_vector_type(4)));
typedef int          i32x8 __attribute__((ext_vector_type(8)));

#define CONV_TM 128
#define CONV_TN 64
#define CONV_TK 32
#define LDSROW  (CONV_TK + 8)   // 40 ushorts = 64B row + 16B pad (TDM pad fields)

__device__ __forceinline__ unsigned short f2bf(float f) {
    unsigned int u = __float_as_uint(f);
    u += 0x7FFFu + ((u >> 16) & 1u);          // round-to-nearest-even
    return (unsigned short)(u >> 16);
}

union Frag16 { unsigned int u[8]; v16bf v; };

// ---------------------------------------------------------------------------
// Implicit-GEMM conv:  M = Cout, N = Ho*Wo (per image), K = Cin*k*k.
// Block tile 128x64, wave tile 32x32 (2x2 WMMA 16x16x32 bf16), batch = grid.z.
// inOff/inCtot and outOff/outCtot implement free slicing / concatenation.
// ---------------------------------------------------------------------------
template<int K, int STRIDE>
__global__ __launch_bounds__(256, 2)
void conv_wmma_bf16(const float* __restrict__ in, int inCtot, int inOff, int Cin,
                    const unsigned short* __restrict__ wq,
                    const float* __restrict__ bias,
                    float* __restrict__ out, int outCtot, int outOff, int Cout,
                    int Hin, int Win, int Hout, int lgW)
{
    constexpr int KK  = K * K;
    constexpr int PAD = (K - 1) / 2;

    __shared__ unsigned short As[2][CONV_TM][LDSROW];
    __shared__ unsigned short Bs[2][CONV_TN][LDSROW];

    const int tid  = threadIdx.x;
    const int lane = tid & 31;
    const int wave = tid >> 5;
    const int half = lane >> 4;
    const int l16  = lane & 15;

    const int Wout = 1 << lgW;
    const int HWo  = Hout << lgW;        // pixels per image
    const int Ktot = Cin * KK;

    const int b    = blockIdx.z;
    const int mblk = blockIdx.y * CONV_TM;
    const int nblk = blockIdx.x * CONV_TN;
    const int wm   = (wave >> 1) * 32;   // wave M offset inside block tile
    const int wn   = (wave & 1) * 32;    // wave N offset inside block tile

    // staging coordinates: each thread owns one k-column, rows step by 8
    const int skk  = tid & 31;
    const int srow = tid >> 5;

    const float* inb = in + (size_t)(b * inCtot + inOff) * Hin * Win;

    // ---- A stage: Tensor Data Mover copies the 128x32 bf16 weight tile ----
    // 2D D#: tile_dim0=32 (k), tile_dim1=128 (rows), row stride = Ktot.
    // pad_interval=3 (16 DWORDs = 64B row), pad_amount=3 (4 DWORDs = 16B)
    // -> LDS pitch of 80B == LDSROW ushorts. OOB rows/cols are zero-filled.
    auto stage_A_tdm = [&](int k0, int buf) {
        unsigned lds = (unsigned)(size_t)(void*)&As[buf][0][0];
        unsigned long long ga =
            (unsigned long long)(size_t)(wq + (size_t)mblk * Ktot + k0);
        u32x4 g0;
        g0[0] = 1u;                                         // count=1 (valid D#)
        g0[1] = lds;                                        // lds_addr
        g0[2] = (unsigned)(ga & 0xFFFFFFFFu);               // global_addr[31:0]
        g0[3] = (unsigned)((ga >> 32) & 0x01FFFFFFu)        // global_addr[56:32]
              | (2u << 30);                                 // type = 2 ("image")
        int td0 = Ktot - k0;                                // valid k in row
        int td1 = Cout - mblk;                              // valid rows
        i32x8 g1;
        g1[0] = (1 << 16)                                   // data_size = 2B
              | (1 << 20)                                   // pad_enable
              | (3 << 22)                                   // pad_interval: 16 dw
              | (3 << 25);                                  // pad_amount: 4 dw
        g1[1] = (td0 & 0xFFFF) << 16;                       // tensor_dim0 lo
        g1[2] = (int)(((unsigned)td0 >> 16) & 0xFFFFu)      // tensor_dim0 hi
              | ((td1 & 0xFFFF) << 16);                     // tensor_dim1 lo
        g1[3] = (int)(((unsigned)td1 >> 16) & 0xFFFFu)      // tensor_dim1 hi
              | (CONV_TK << 16);                            // tile_dim0 = 32
        g1[4] = CONV_TM;                                    // tile_dim1 = 128
        g1[5] = Ktot;                                       // dim0_stride lo
        g1[6] = 0;
        g1[7] = 0;
        i32x4 z4 = {0, 0, 0, 0};
#if __clang_major__ >= 23
        i32x8 z8 = {0, 0, 0, 0, 0, 0, 0, 0};
        __builtin_amdgcn_tensor_load_to_lds(g0, g1, z4, z4, z8, 0);
#else
        __builtin_amdgcn_tensor_load_to_lds(g0, g1, z4, z4, 0);
#endif
    };

    // ---- B stage: im2col activation tile [64 n x 32 k], fp32 -> bf16 ----
    auto stage_B = [&](int k0, int buf) {
        const int gk = k0 + skk;
        const bool kval = gk < Ktot;
        // K-decomposition once per chunk (compile-time div/mod -> mul+shift)
        int cin = gk / KK;
        int rem = gk - cin * KK;
        int kh  = rem / K;
        int kw  = rem - kh * K;
        const float* inc = inb + (size_t)cin * Hin * Win;
        #pragma unroll
        for (int it = 0; it < CONV_TN / 8; ++it) {
            int n  = it * 8 + srow;
            int pr = nblk + n;
            unsigned short v = 0;
            if (pr < HWo && kval) {
                int ho = pr >> lgW;
                int wo = pr & (Wout - 1);
                int hi = ho * STRIDE - PAD + kh;
                int wi = wo * STRIDE - PAD + kw;
                if (hi >= 0 && hi < Hin && wi >= 0 && wi < Win)
                    v = f2bf(inc[hi * Win + wi]);
            }
            Bs[buf][n][skk] = v;
        }
    };

    v8f acc[2][2] = {};

    const int nchunks = (Ktot + CONV_TK - 1) / CONV_TK;
    if (wave == 0) stage_A_tdm(0, 0);
    stage_B(0, 0);
    if (wave == 0) __builtin_amdgcn_s_wait_tensorcnt(0);
    __syncthreads();

    for (int i = 0; i < nchunks; ++i) {
        const int buf = i & 1;
        if (i + 1 < nchunks) {
            if (wave == 0) stage_A_tdm((i + 1) * CONV_TK, buf ^ 1);  // DMA
            stage_B((i + 1) * CONV_TK, buf ^ 1);                     // vmem
        }

        // ---- fragment assembly per CDNA5 ISA 16-bit layouts (§7.12.2) ----
        Frag16 a0, a1, b0, b1;
        #pragma unroll
        for (int j = 0; j < 8; ++j) {
            int ka = (j < 4 ? 2 * j : 16 + 2 * (j - 4)) + 8 * half;   // A 16x32
            a0.u[j] = *(const unsigned int*)&As[buf][wm + l16][ka];
            a1.u[j] = *(const unsigned int*)&As[buf][wm + 16 + l16][ka];
            int kb = (j < 4 ? 2 * j : 16 + 2 * (j - 4)) + 8 * half;   // B 32x16
            b0.u[j] = *(const unsigned int*)&Bs[buf][wn + l16][kb];
            b1.u[j] = *(const unsigned int*)&Bs[buf][wn + 16 + l16][kb];
        }
        acc[0][0] = __builtin_amdgcn_wmma_f32_16x16x32_bf16(false, a0.v, false, b0.v, (short)0, acc[0][0], false, false);
        acc[0][1] = __builtin_amdgcn_wmma_f32_16x16x32_bf16(false, a0.v, false, b1.v, (short)0, acc[0][1], false, false);
        acc[1][0] = __builtin_amdgcn_wmma_f32_16x16x32_bf16(false, a1.v, false, b0.v, (short)0, acc[1][0], false, false);
        acc[1][1] = __builtin_amdgcn_wmma_f32_16x16x32_bf16(false, a1.v, false, b1.v, (short)0, acc[1][1], false, false);

        if (wave == 0 && i + 1 < nchunks) __builtin_amdgcn_s_wait_tensorcnt(0);
        __syncthreads();
    }

    // ---- epilogue: bias add, scatter to NCHW (flat pixel index) ----
    #pragma unroll
    for (int ni = 0; ni < 2; ++ni) {
        int pr = nblk + wn + ni * 16 + l16;      // C/D layout: N = lane&15
        if (pr >= HWo) continue;
        #pragma unroll
        for (int mi = 0; mi < 2; ++mi) {
            #pragma unroll
            for (int r = 0; r < 8; ++r) {
                int m = mblk + wm + mi * 16 + r + 8 * half;  // M = vgpr + 8*half
                if (m < Cout)
                    out[(size_t)(b * outCtot + outOff + m) * HWo + pr] =
                        acc[mi][ni][r] + bias[m];
            }
        }
    }
}

// ---------------------------------------------------------------------------
// BatchNorm statistics: one block per channel, LDS tree reduction.
// ---------------------------------------------------------------------------
__global__ __launch_bounds__(256)
void bn_stats_k(const float* __restrict__ y, int Ctot, int off, int C,
                int B, int HW, float* __restrict__ meanvar)
{
    int c = blockIdx.x;
    float s = 0.f, s2 = 0.f;
    int total = B * HW;
    for (int i = threadIdx.x; i < total; i += blockDim.x) {
        int b = i / HW, p = i - b * HW;
        float v = y[((b * Ctot) + off + c) * HW + p];
        s += v; s2 += v * v;
    }
    __shared__ float r1[256], r2[256];
    r1[threadIdx.x] = s; r2[threadIdx.x] = s2;
    __syncthreads();
    for (int o = 128; o > 0; o >>= 1) {
        if ((int)threadIdx.x < o) { r1[threadIdx.x] += r1[threadIdx.x + o];
                                    r2[threadIdx.x] += r2[threadIdx.x + o]; }
        __syncthreads();
    }
    if (threadIdx.x == 0) {
        float n = (float)total;
        float m = r1[0] / n;
        meanvar[2 * c]     = m;
        meanvar[2 * c + 1] = r2[0] / n - m * m;
    }
}

// ---------------------------------------------------------------------------
// Fused BN apply + ReLU (in place).
// ---------------------------------------------------------------------------
__global__ __launch_bounds__(256)
void bn_apply_relu_k(float* __restrict__ y, int Ctot, int off, int C,
                     int B, int HW,
                     const float* __restrict__ g, const float* __restrict__ bt,
                     const float* __restrict__ meanvar)
{
    int idx = blockIdx.x * blockDim.x + threadIdx.x;
    int total = B * C * HW;
    if (idx >= total) return;
    int b = idx / (C * HW), r = idx - b * C * HW;
    int c = r / HW, p = r - c * HW;
    float* ptr = y + ((b * Ctot) + off + c) * HW + p;
    float m = meanvar[2 * c], var = meanvar[2 * c + 1];
    float v = (*ptr - m) * __frsqrt_rn(var + 1e-5f) * g[c] + bt[c];
    *ptr = fmaxf(v, 0.f);
}

// ---------------------------------------------------------------------------
// Correlation (MAX_D=20, stride-2 -> 21x21 = 441 displacements) + leaky ReLU.
// Writes channels [32, 473) of the conv3_1 input buffer (redir occupies 0..31).
// ---------------------------------------------------------------------------
__global__ __launch_bounds__(256)
void corr_leaky_k(const float* __restrict__ L, const float* __restrict__ R,
                  float* __restrict__ out473, int B, int C, int H, int W)
{
    int p = blockIdx.x * blockDim.x + threadIdx.x;
    int d = blockIdx.y;                       // 0..440
    int HW = H * W;
    if (p >= B * HW) return;
    int b = p / HW, r = p - b * HW;
    int h = r / W, w = r - h * W;
    int di = d / 21, dj = d - di * 21;
    int dx = -20 + 2 * di;                    // first meshgrid axis
    int dy = -20 + 2 * dj;
    int hs = h - dy, ws = w - dx;
    float s = 0.f;
    if (hs >= 0 && hs < H && ws >= 0 && ws < W) {
        const float* lp = L + (size_t)b * C * HW + h  * W + w;
        const float* rp = R + (size_t)b * C * HW + hs * W + ws;
        for (int c = 0; c < C; ++c) s += lp[c * HW] * rp[c * HW];
    }
    s = (s > 0.f) ? s : 0.1f * s;
    out473[((size_t)b * 473 + 32 + d) * HW + h * W + w] = s;
}

// ---------------------------------------------------------------------------
// fp32 -> bf16 weight quantizer.
// ---------------------------------------------------------------------------
__global__ __launch_bounds__(256)
void cvt_bf16_k(const float* __restrict__ in, unsigned short* __restrict__ out, int n)
{
    int i = blockIdx.x * blockDim.x + threadIdx.x;
    if (i < n) out[i] = f2bf(in[i]);
}

// ===========================================================================
// Host orchestration.
// ===========================================================================
namespace {
struct P { int iw, ib, ig, ibt; int Co, Ci, K; };
}

extern "C" void kernel_launch(void* const* d_in, const int* in_sizes, int n_in,
                              void* d_out, int out_size, void* d_ws, size_t ws_size,
                              hipStream_t stream)
{
    (void)in_sizes; (void)n_in; (void)out_size; (void)ws_size;
    const int B = 4;
    const float* x = (const float*)d_in[0];

    // Param tensors in setup_inputs() dict order (w,b[,g,bt] per layer).
    const P conv1  = { 1,  2,  3,  4,    64,    3, 7};
    const P conv2  = { 5,  6,  7,  8,   128,   64, 5};
    const P pf2    = { 9, 10, -1, -1,     2,  128, 3};
    const P conv3  = {11, 12, 13, 14,   256,  130, 5};
    const P pf3    = {15, 16, -1, -1,     2,  256, 3};
    const P redir  = {17, 18, 19, 20,    32,  258, 1};
    const P conv31 = {21, 22, 23, 24,   256,  473, 3};
    const P conv4  = {25, 26, 27, 28,   512,  256, 3};
    const P conv41 = {29, 30, 31, 32,   512,  512, 3};
    const P pf4    = {33, 34, -1, -1,     2,  512, 3};
    const P conv5  = {35, 36, 37, 38,   514,  514, 3};
    const P conv51 = {39, 40, 41, 42,   514,  514, 3};
    const P pf5    = {43, 44, -1, -1,     2,  514, 3};
    const P conv6  = {45, 46, 47, 48,  1024,  516, 3};
    const P conv61 = {49, 50, 51, 52,  1024, 1024, 3};
    const P pf6    = {53, 54, -1, -1,     2, 1024, 3};

    // ---- workspace carving ----
    size_t cur = 0;
    auto alloc_f = [&](size_t n) {
        float* p = (float*)((char*)d_ws + cur);
        cur += n * sizeof(float);
        return p;
    };
    auto alloc_h = [&](size_t n) {
        unsigned short* p = (unsigned short*)((char*)d_ws + cur);
        cur += ((n + 1) & ~(size_t)1) * sizeof(unsigned short);
        return p;
    };

    // bf16 weight copies
    const P specs[16] = {conv1, conv2, pf2, conv3, pf3, redir, conv31, conv4,
                         conv41, pf4, conv5, conv51, pf5, conv6, conv61, pf6};
    unsigned short* wq[16];
    for (int i = 0; i < 16; ++i) {
        size_t n = (size_t)specs[i].Co * specs[i].Ci * specs[i].K * specs[i].K;
        wq[i] = alloc_h(n);
        cvt_bf16_k<<<(int)((n + 255) / 256), 256, 0, stream>>>(
            (const float*)d_in[specs[i].iw], wq[i], (int)n);
    }
    enum { Q1, Q2, QP2, Q3, QP3, QR, Q31, Q4, Q41, QP4, Q5, Q51, QP5, Q6, Q61, QP6 };

    // activation scratch
    float* c1     = alloc_f((size_t)B * 64   * 192 * 256);
    float* c2b    = alloc_f((size_t)B * 130  *  96 * 128);
    float* c3a    = alloc_f((size_t)B * 258  *  48 *  64);
    float* c3b    = alloc_f((size_t)B * 258  *  48 *  64);
    float* corrin = alloc_f((size_t)B * 473  *  48 *  64);
    float* c4pre  = alloc_f((size_t)B * 512  *  24 *  32);
    float* c5pre  = alloc_f((size_t)B * 514  *  12 *  16);
    float* c6pre  = alloc_f((size_t)B * 1024 *   6 *   8);
    float* meanvar = alloc_f(2048 * 2);

    // outputs packed directly into d_out (c2a, c31, c4, c5, c6)
    float* O    = (float*)d_out;
    float* c2a  = O;                                        // 4*130*96*128
    float* c31o = c2a  + (size_t)B * 130 * 96 * 128;        // 4*256*48*64
    float* c4o  = c31o + (size_t)B * 256 * 48 * 64;         // 4*514*24*32
    float* c5o  = c4o  + (size_t)B * 514 * 24 * 32;         // 4*516*12*16
    float* c6o  = c5o  + (size_t)B * 516 * 12 * 16;         // 4*1026*6*8

    // ---- launch helpers ----
    auto conv = [&](const float* in, int inCtot, int inOff, const P& p,
                    const unsigned short* w, float* out, int outCtot, int outOff,
                    int Hin, int Win, int Hout, int Wout, int stride) {
        int lgW = 31 - __builtin_clz((unsigned)Wout);  // Wout is a power of two
        int HWo = Hout * Wout;
        dim3 grid((HWo + CONV_TN - 1) / CONV_TN, (p.Co + CONV_TM - 1) / CONV_TM, B);
        const float* bias = (const float*)d_in[p.ib];
        if (p.K == 7)
            conv_wmma_bf16<7, 2><<<grid, 256, 0, stream>>>(
                in, inCtot, inOff, p.Ci, w, bias, out, outCtot, outOff, p.Co,
                Hin, Win, Hout, lgW);
        else if (p.K == 5)
            conv_wmma_bf16<5, 2><<<grid, 256, 0, stream>>>(
                in, inCtot, inOff, p.Ci, w, bias, out, outCtot, outOff, p.Co,
                Hin, Win, Hout, lgW);
        else if (p.K == 3 && stride == 2)
            conv_wmma_bf16<3, 2><<<grid, 256, 0, stream>>>(
                in, inCtot, inOff, p.Ci, w, bias, out, outCtot, outOff, p.Co,
                Hin, Win, Hout, lgW);
        else if (p.K == 3)
            conv_wmma_bf16<3, 1><<<grid, 256, 0, stream>>>(
                in, inCtot, inOff, p.Ci, w, bias, out, outCtot, outOff, p.Co,
                Hin, Win, Hout, lgW);
        else
            conv_wmma_bf16<1, 1><<<grid, 256, 0, stream>>>(
                in, inCtot, inOff, p.Ci, w, bias, out, outCtot, outOff, p.Co,
                Hin, Win, Hout, lgW);
    };
    auto bn = [&](float* y, int Ctot, int off, const P& p, int HW) {
        bn_stats_k<<<p.Co, 256, 0, stream>>>(y, Ctot, off, p.Co, B, HW, meanvar);
        int tot = B * p.Co * HW;
        bn_apply_relu_k<<<(tot + 255) / 256, 256, 0, stream>>>(
            y, Ctot, off, p.Co, B, HW,
            (const float*)d_in[p.ig], (const float*)d_in[p.ibt], meanvar);
    };

    // ---- shared branch: x[:,off:off+3] -> (c2 concat, c3 concat) ----
    auto branch = [&](int xoff, float* c2dst, float* c3dst) {
        conv(x, 6, xoff, conv1, wq[Q1], c1, 64, 0, 384, 512, 192, 256, 2);
        bn(c1, 64, 0, conv1, 192 * 256);
        conv(c1, 64, 0, conv2, wq[Q2], c2dst, 130, 0, 192, 256, 96, 128, 2);
        bn(c2dst, 130, 0, conv2, 96 * 128);
        conv(c2dst, 130, 0, pf2, wq[QP2], c2dst, 130, 128, 96, 128, 96, 128, 1);
        conv(c2dst, 130, 0, conv3, wq[Q3], c3dst, 258, 0, 96, 128, 48, 64, 2);
        bn(c3dst, 258, 0, conv3, 48 * 64);
        conv(c3dst, 258, 0, pf3, wq[QP3], c3dst, 258, 256, 48, 64, 48, 64, 1);
    };

    branch(0, c2a, c3a);     // left image -> c2a goes straight to d_out
    branch(3, c2b, c3b);     // right image

    // conv_redir (1x1) -> corrin ch[0..31]; correlation -> ch[32..472]
    conv(c3a, 258, 0, redir, wq[QR], corrin, 473, 0, 48, 64, 48, 64, 1);
    bn(corrin, 473, 0, redir, 48 * 64);
    {
        int pix = B * 48 * 64;
        dim3 grid((pix + 255) / 256, 441);
        corr_leaky_k<<<grid, 256, 0, stream>>>(c3a, c3b, corrin, B, 258, 48, 64);
    }

    // conv3_1 -> c31 output
    conv(corrin, 473, 0, conv31, wq[Q31], c31o, 256, 0, 48, 64, 48, 64, 1);
    bn(c31o, 256, 0, conv31, 48 * 64);

    // conv4 / conv4_1 / pf4 -> c4 output (514 ch)
    conv(c31o, 256, 0, conv4, wq[Q4], c4pre, 512, 0, 48, 64, 24, 32, 2);
    bn(c4pre, 512, 0, conv4, 24 * 32);
    conv(c4pre, 512, 0, conv41, wq[Q41], c4o, 514, 0, 24, 32, 24, 32, 1);
    bn(c4o, 514, 0, conv41, 24 * 32);
    conv(c4o, 514, 0, pf4, wq[QP4], c4o, 514, 512, 24, 32, 24, 32, 1);

    // conv5 / conv5_1 / pf5 -> c5 output (516 ch)
    conv(c4o, 514, 0, conv5, wq[Q5], c5pre, 514, 0, 24, 32, 12, 16, 2);
    bn(c5pre, 514, 0, conv5, 12 * 16);
    conv(c5pre, 514, 0, conv51, wq[Q51], c5o, 516, 0, 12, 16, 12, 16, 1);
    bn(c5o, 516, 0, conv51, 12 * 16);
    conv(c5o, 516, 0, pf5, wq[QP5], c5o, 516, 514, 12, 16, 12, 16, 1);

    // conv6 / conv6_1 / pf6 -> c6 output (1026 ch)
    conv(c5o, 516, 0, conv6, wq[Q6], c6pre, 1024, 0, 12, 16, 6, 8, 2);
    bn(c6pre, 1024, 0, conv6, 6 * 8);
    conv(c6pre, 1024, 0, conv61, wq[Q61], c6o, 1026, 0, 6, 8, 6, 8, 1);
    bn(c6o, 1026, 0, conv61, 6 * 8);
    conv(c6o, 1026, 0, pf6, wq[QP6], c6o, 1026, 1024, 6, 8, 6, 8, 1);
}
